// Decoder_78924319031620
// MI455X (gfx1250) — matrix operand order repaired and checked by reading, not compiled, not run on hardware
//
#include <hip/hip_runtime.h>

// ---------------- problem constants ----------------
#define D_MODEL 512
#define SEQ     1024
#define BATCH   8
#define HEADS   8
#define DEPTH   64
#define RELN    513          // 2*MAX_REL+1
#define RELP    528          // padded
#define FFN     2048
#define NROWS   (BATCH*SEQ)  // 8192

#ifndef __has_builtin
#define __has_builtin(x) 0
#endif
#if __has_builtin(__builtin_amdgcn_global_load_async_to_lds_b128) && \
    __has_builtin(__builtin_amdgcn_s_wait_asynccnt)
#define USE_ASYNC_STAGE 1
#else
#define USE_ASYNC_STAGE 0
#endif

typedef unsigned short u16;
typedef __bf16 bf16_t;
typedef bf16_t v16bf __attribute__((ext_vector_type(16)));
typedef float  v8f   __attribute__((ext_vector_type(8)));
typedef float  f32x4 __attribute__((ext_vector_type(4)));
typedef unsigned int u32x4 __attribute__((ext_vector_type(4)));
typedef u16 u16x8 __attribute__((ext_vector_type(8)));
typedef u16 u16x4 __attribute__((ext_vector_type(4)));
#if USE_ASYNC_STAGE
// exact parameter types per hipcc diagnostic: vector_size(16) int, AS1 / AS3
typedef int v4i __attribute__((vector_size(4 * sizeof(int))));
typedef __attribute__((address_space(1))) v4i as1_v4i;
typedef __attribute__((address_space(3))) v4i as3_v4i;
#endif

union FragCast { u32x4 u[2]; v16bf v; };
union BfBits { float f; unsigned int u; };

__device__ __forceinline__ u16 f2bf(float f) {
  BfBits x; x.f = f;
  unsigned int u = x.u + 0x7fffu + ((x.u >> 16) & 1u);  // RNE
  return (u16)(u >> 16);
}
__device__ __forceinline__ float bf2f(u16 b) {
  BfBits x; x.u = ((unsigned int)b) << 16;
  return x.f;
}
__device__ __forceinline__ v16bf ld_frag(const u16* p0, const u16* p1) {
  FragCast f;
  f.u[0] = *reinterpret_cast<const u32x4*>(p0);   // ds_load_b128
  f.u[1] = *reinterpret_cast<const u32x4*>(p1);   // ds_load_b128
  return f.v;
}

// one K=32 step of the 32x32 wave tile (4 x v_wmma_f32_16x16x32_bf16)
__device__ __forceinline__ void compute_step(const u16 (*Asb)[40], const u16 (*Bsb)[40],
                                             int wm, int wn, int lr, int half,
                                             v8f (&acc)[2][2])
{
  v16bf af0 = ld_frag(&Asb[wm*32 + lr][half*8],      &Asb[wm*32 + lr][half*8 + 16]);
  v16bf af1 = ld_frag(&Asb[wm*32 + 16 + lr][half*8], &Asb[wm*32 + 16 + lr][half*8 + 16]);
  v16bf bf0 = ld_frag(&Bsb[wn*32 + lr][half*8],      &Bsb[wn*32 + lr][half*8 + 16]);
  v16bf bf1 = ld_frag(&Bsb[wn*32 + 16 + lr][half*8], &Bsb[wn*32 + 16 + lr][half*8 + 16]);
  acc[0][0] = __builtin_amdgcn_wmma_f32_16x16x32_bf16(false, af0, false, bf0,
                                                      (short)0, acc[0][0], false, false);
  acc[0][1] = __builtin_amdgcn_wmma_f32_16x16x32_bf16(false, af0, false, bf1,
                                                      (short)0, acc[0][1], false, false);
  acc[1][0] = __builtin_amdgcn_wmma_f32_16x16x32_bf16(false, af1, false, bf0,
                                                      (short)0, acc[1][0], false, false);
  acc[1][1] = __builtin_amdgcn_wmma_f32_16x16x32_bf16(false, af1, false, bf1,
                                                      (short)0, acc[1][1], false, false);
}

// ---------------------------------------------------------------------------
// bf16 WMMA GEMM: C = A[M][K] @ Bt[N][K]^T. Block tile 128x64, 8 wave32,
// each wave 32x32. Staging: async-to-LDS double buffer (gfx1250) or
// register-pipelined b128 copies as fallback.
//   mode 0: linear z-batch; epilogue bias/relu/resid -> C fp32 / Cbf bf16 /
//           Ctv bf16 (V transposed [b,h][d][q])
//   mode 1: attention scores: (A@Bt^T + relq)/8 + mask*-1e9 -> C (per b,h)
//   mode 2: context: per (b,h), A=attn_w bf16, Bt=V^T slice -> Cbf
// ---------------------------------------------------------------------------
__global__ __launch_bounds__(256)
void wmma_gemm(int mode, int Nreal, int K,
               int lda, int ldb, int ldc,
               const u16* __restrict__ A, const u16* __restrict__ Bt,
               float* __restrict__ C, u16* __restrict__ Cbf,
               u16* __restrict__ Ctv,
               const float* __restrict__ bias, const float* __restrict__ resid,
               int relu,
               long long aStrideZ, long long cStrideZ,
               const u16* __restrict__ relq, const float* __restrict__ mask)
{
#if USE_ASYNC_STAGE
  __shared__ __align__(16) u16 As[2][128][40];
  __shared__ __align__(16) u16 Bs[2][64][40];
#else
  __shared__ __align__(16) u16 As[1][128][40];
  __shared__ __align__(16) u16 Bs[1][64][40];
#endif

  const int tid = threadIdx.x;
  const int z   = blockIdx.z;

  if (mode == 1) {                       // scores: z = b*8+h
    int b = z >> 3, h = z & 7;
    A    += (size_t)b * SEQ * D_MODEL + h * DEPTH;   // Q bf16 head slice
    Bt   += (size_t)b * SEQ * D_MODEL + h * DEPTH;   // K bf16 head slice
    C    += (size_t)z * SEQ * SEQ;                   // attn_w[b,h] fp32
    relq += ((size_t)b * SEQ * HEADS + h) * RELP;
    mask += (size_t)b * SEQ * SEQ;
  } else if (mode == 2) {                // context: z = b*8+h
    int b = z >> 3, h = z & 7;
    A   += (size_t)z * SEQ * SEQ;                    // attn_w bf16
    Bt  += (size_t)z * DEPTH * SEQ;                  // V^T slice [64][1024]
    Cbf += (size_t)b * SEQ * D_MODEL + h * DEPTH;    // ctx bf16 slice
  } else {
    A += aStrideZ * (long long)z;
    if (C)   C   += cStrideZ * (long long)z;
    if (Cbf) Cbf += cStrideZ * (long long)z;
    if (Ctv) Ctv += cStrideZ * (long long)z;
  }

  const int m0 = blockIdx.y * 128;
  const int n0 = blockIdx.x * 64;
  const u16* Ag = A + (size_t)m0 * lda;

  const int wid  = tid >> 5;
  const int lane = tid & 31;
  const int wm   = wid >> 1;    // 0..3 : 32-row subtile
  const int wn   = wid & 1;     // 0..1 : 32-col subtile
  const int half = lane >> 4;
  const int lr   = lane & 15;

  // staging geometry: A tile 128x32 = 512 x 16B chunks = 2/thread;
  // B tile 64x32 = 256 chunks = 1/thread. Pure b128 moves.
  const int rA0 = tid >> 2,         cA0 = (tid & 3) * 8;
  const int rA1 = (tid + 256) >> 2, cA1 = ((tid + 256) & 3) * 8;
  const int rB  = tid >> 2,         cB  = (tid & 3) * 8;
  // OOB B rows clamp to a valid row: they feed only output columns that the
  // epilogue never stores (gn < Nreal guard), so the values are irrelevant.
  const int bRow = (n0 + rB < Nreal) ? (n0 + rB) : (Nreal - 1);
  const size_t aOff0 = (size_t)rA0 * lda + cA0;
  const size_t aOff1 = (size_t)rA1 * lda + cA1;
  const size_t bOff  = (size_t)bRow * ldb + cB;

  v8f acc[2][2];
  #pragma unroll
  for (int i = 0; i < 8; ++i) {
    acc[0][0][i] = 0.f; acc[0][1][i] = 0.f;
    acc[1][0][i] = 0.f; acc[1][1][i] = 0.f;
  }

#if USE_ASYNC_STAGE
  // -------- async-to-LDS double-buffered staging (ASYNCcnt) --------
  auto issue = [&](int buf, int k0) {
    __builtin_amdgcn_global_load_async_to_lds_b128(
        (as1_v4i*)(Ag + aOff0 + k0), (as3_v4i*)&As[buf][rA0][cA0], 0, 0);
    __builtin_amdgcn_global_load_async_to_lds_b128(
        (as1_v4i*)(Ag + aOff1 + k0), (as3_v4i*)&As[buf][rA1][cA1], 0, 0);
    __builtin_amdgcn_global_load_async_to_lds_b128(
        (as1_v4i*)(Bt + bOff + k0),  (as3_v4i*)&Bs[buf][rB][cB],   0, 0);
  };
  issue(0, 0);
  int buf = 0;
  for (int k0 = 0; k0 < K; k0 += 32) {
    const int kn = k0 + 32;
    if (kn < K) {
      issue(buf ^ 1, kn);                    // overlap next tile
      __builtin_amdgcn_s_wait_asynccnt(3);   // current tile's 3 ops done
    } else {
      __builtin_amdgcn_s_wait_asynccnt(0);
    }
    __syncthreads();                         // whole tile visible to all waves
    compute_step(As[buf], Bs[buf], wm, wn, lr, half, acc);
    __syncthreads();                         // done reading before next overwrite
    buf ^= 1;
  }
#else
  // -------- register-pipelined staging fallback --------
  u16x8 ra0 = *(const u16x8*)&Ag[aOff0];
  u16x8 ra1 = *(const u16x8*)&Ag[aOff1];
  u16x8 rb  = *(const u16x8*)&Bt[bOff];
  for (int k0 = 0; k0 < K; k0 += 32) {
    *(u16x8*)&As[0][rA0][cA0] = ra0;
    *(u16x8*)&As[0][rA1][cA1] = ra1;
    *(u16x8*)&Bs[0][rB][cB]   = rb;
    __syncthreads();
    const int kn = k0 + 32;
    if (kn < K) {
      ra0 = *(const u16x8*)&Ag[aOff0 + kn];
      ra1 = *(const u16x8*)&Ag[aOff1 + kn];
      rb  = *(const u16x8*)&Bt[bOff + kn];
      if (kn + 32 < K)
        __builtin_prefetch(&Ag[aOff0 + kn + 32], 0, 1);
    }
    compute_step(As[0], Bs[0], wm, wn, lr, half, acc);
    __syncthreads();
  }
#endif

  // D layout: VGPR i -> row = i + 8*half, col = lane&15
  if (mode == 1) {
    #pragma unroll
    for (int mi = 0; mi < 2; ++mi) {
      #pragma unroll
      for (int ni = 0; ni < 2; ++ni) {
        int gk = n0 + wn*32 + ni*16 + lr;
        #pragma unroll
        for (int i = 0; i < 8; ++i) {
          int gq = m0 + wm*32 + mi*16 + (half << 3) + i;
          int r  = gk - gq;
          r = r < -256 ? -256 : (r > 256 ? 256 : r);
          float v = (acc[mi][ni][i] + bf2f(relq[(size_t)gq*(HEADS*RELP) + r + 256]))
                      * 0.125f
                  + mask[(size_t)gq * SEQ + gk] * -1.0e9f;
          C[(size_t)gq * ldc + gk] = v;
        }
      }
    }
  } else {
    #pragma unroll
    for (int mi = 0; mi < 2; ++mi) {
      #pragma unroll
      for (int ni = 0; ni < 2; ++ni) {
        int gn = n0 + wn*32 + ni*16 + lr;
        if (gn < Nreal) {
          #pragma unroll
          for (int i = 0; i < 8; ++i) {
            int gm = m0 + wm*32 + mi*16 + (half << 3) + i;
            float v = acc[mi][ni][i];
            if (bias)  v += bias[gn];
            if (relu)  v  = fmaxf(v, 0.f);
            if (resid) v += resid[(size_t)gm * ldc + gn];
            if (C)   C[(size_t)gm * ldc + gn]   = v;
            if (Cbf) Cbf[(size_t)gm * ldc + gn] = f2bf(v);
            if (Ctv) {  // V^T layout [b,h][d][q]
              int b = gm >> 10, q = gm & 1023, h = gn >> 6, d = gn & 63;
              Ctv[(((size_t)(b*HEADS + h) * DEPTH) + d) * SEQ + q] = f2bf(v);
            }
          }
        }
      }
    }
  }
}

// ---------------- softmax over 1024-rows, fp32 in place + bf16 copy --------
__global__ __launch_bounds__(256)
void softmax1024(float* __restrict__ w, u16* __restrict__ wbf)
{
  __shared__ float red[256];
  const size_t row = blockIdx.x;
  float* p = w + row * 1024;
  u16*   q = wbf + row * 1024;
  const int t = threadIdx.x;
  float v0 = p[t], v1 = p[t+256], v2 = p[t+512], v3 = p[t+768];
  float m = fmaxf(fmaxf(v0, v1), fmaxf(v2, v3));
  red[t] = m; __syncthreads();
  for (int s = 128; s > 0; s >>= 1) {
    if (t < s) red[t] = fmaxf(red[t], red[t+s]);
    __syncthreads();
  }
  m = red[0]; __syncthreads();
  v0 = __expf(v0 - m); v1 = __expf(v1 - m);
  v2 = __expf(v2 - m); v3 = __expf(v3 - m);
  red[t] = v0 + v1 + v2 + v3; __syncthreads();
  for (int s = 128; s > 0; s >>= 1) {
    if (t < s) red[t] += red[t+s];
    __syncthreads();
  }
  float inv = 1.0f / red[0];
  v0 *= inv; v1 *= inv; v2 *= inv; v3 *= inv;
  p[t] = v0; p[t+256] = v1; p[t+512] = v2; p[t+768] = v3;
  q[t] = f2bf(v0); q[t+256] = f2bf(v1); q[t+512] = f2bf(v2); q[t+768] = f2bf(v3);
}

// ---------------- layernorm rows of 512: fp32 out + optional bf16 ----------
__global__ __launch_bounds__(256)
void layernorm512(const float* __restrict__ x, const float* __restrict__ g,
                  const float* __restrict__ bta, float* __restrict__ y,
                  u16* __restrict__ ybf)
{
  __shared__ float rs[256], rq[256];
  const size_t row = blockIdx.x;
  const float* p = x + row * D_MODEL;
  const int t = threadIdx.x;
  float a = p[t], b = p[t+256];
  rs[t] = a + b; rq[t] = a*a + b*b;
  __syncthreads();
  for (int s = 128; s > 0; s >>= 1) {
    if (t < s) { rs[t] += rs[t+s]; rq[t] += rq[t+s]; }
    __syncthreads();
  }
  float mean = rs[0] * (1.f/512.f);
  float var  = rq[0] * (1.f/512.f) - mean*mean;
  float rstd = rsqrtf(var + 1e-3f);
  float o0 = (a - mean) * rstd * g[t]     + bta[t];
  float o1 = (b - mean) * rstd * g[t+256] + bta[t+256];
  float* q = y + row * D_MODEL;
  q[t] = o0; q[t+256] = o1;
  if (ybf) { u16* qb = ybf + row * D_MODEL; qb[t] = f2bf(o0); qb[t+256] = f2bf(o1); }
}

// ---------------- fp32 -> bf16 straight convert (vectorized) ---------------
__global__ __launch_bounds__(256)
void cvt_bf16(const float* __restrict__ x, u16* __restrict__ y, int n4)
{
  int i = blockIdx.x * 256 + threadIdx.x;
  if (i < n4) {
    f32x4 v = *(const f32x4*)(x + (size_t)i * 4);
    u16x4 r;
    r[0] = f2bf(v[0]); r[1] = f2bf(v[1]); r[2] = f2bf(v[2]); r[3] = f2bf(v[3]);
    *(u16x4*)(y + (size_t)i * 4) = r;
  }
}

// ---------------- fp32 W[K][N] -> bf16 Wt[N][K] (K,N multiples of 32) ------
__global__ __launch_bounds__(256)
void cvt_t_bf16(const float* __restrict__ W, u16* __restrict__ Wt, int K, int N)
{
  __shared__ float t[32][33];
  const int n0 = blockIdx.x * 32, k0 = blockIdx.y * 32;
  const int tx = threadIdx.x & 31, ty = threadIdx.x >> 5;
  #pragma unroll
  for (int i = 0; i < 4; ++i) {
    int k = ty + i * 8;
    t[k][tx] = W[(size_t)(k0 + k) * N + n0 + tx];
  }
  __syncthreads();
  #pragma unroll
  for (int i = 0; i < 4; ++i) {
    int n = ty + i * 8;
    Wt[(size_t)(n0 + n) * K + k0 + tx] = f2bf(t[tx][n]);
  }
}

// ---------------- host-side launch helper ----------------
static void launch_gemm(hipStream_t s, int mode, int M, int N, int Nreal, int K,
                        int lda, int ldb, int ldc,
                        const u16* A, const u16* Bt,
                        float* C, u16* Cbf, u16* Ctv,
                        const float* bias, const float* resid, int relu,
                        long long az, long long cz,
                        const u16* relq, const float* mask, int batch)
{
  dim3 grid((N + 63) / 64, M / 128, batch);
  wmma_gemm<<<grid, dim3(256), 0, s>>>(mode, Nreal, K, lda, ldb, ldc,
                                       A, Bt, C, Cbf, Ctv, bias, resid, relu,
                                       az, cz, relq, mask);
}

extern "C" void kernel_launch(void* const* d_in, const int* in_sizes, int n_in,
                              void* d_out, int out_size, void* d_ws, size_t ws_size,
                              hipStream_t stream)
{
  (void)in_sizes; (void)n_in; (void)out_size; (void)ws_size;

  const float* inputs    = (const float*)d_in[0];
  const float* enc_kv    = (const float*)d_in[1];
  const float* pad_mask  = (const float*)d_in[2];
  const float* look_mask = (const float*)d_in[3];
  const float* sa_Wq = (const float*)d_in[5];  const float* sa_bq = (const float*)d_in[6];
  const float* sa_Wk = (const float*)d_in[7];  const float* sa_bk = (const float*)d_in[8];
  const float* sa_Wv = (const float*)d_in[9];  const float* sa_bv = (const float*)d_in[10];
  const float* sa_Wo = (const float*)d_in[11]; const float* sa_bo = (const float*)d_in[12];
  const float* sa_rel = (const float*)d_in[13];
  const float* sa_g = (const float*)d_in[14];  const float* sa_b = (const float*)d_in[15];
  const float* ca_Wq = (const float*)d_in[16]; const float* ca_bq = (const float*)d_in[17];
  const float* ca_Wk = (const float*)d_in[18]; const float* ca_bk = (const float*)d_in[19];
  const float* ca_Wv = (const float*)d_in[20]; const float* ca_bv = (const float*)d_in[21];
  const float* ca_Wo = (const float*)d_in[22]; const float* ca_bo = (const float*)d_in[23];
  const float* ca_rel = (const float*)d_in[24];
  const float* ca_g = (const float*)d_in[25];  const float* ca_b = (const float*)d_in[26];
  const float* W1 = (const float*)d_in[27];    const float* b1 = (const float*)d_in[28];
  const float* W2 = (const float*)d_in[29];    const float* b2 = (const float*)d_in[30];
  const float* fg = (const float*)d_in[31];    const float* fb = (const float*)d_in[32];

  float* out      = (float*)d_out;
  float* attnw_sa = out + (size_t)NROWS * D_MODEL;
  float* attnw_ca = attnw_sa + (size_t)BATCH * HEADS * SEQ * SEQ;

  // ---- workspace carve (bytes, 256B aligned chunks); total ~360 MB ----
  char* w = (char*)d_ws; size_t off = 0;
  auto carve = [&](size_t bytes) { void* p = w + off; off += (bytes + 255) & ~(size_t)255; return p; };
  const size_t ND = (size_t)NROWS * D_MODEL;
  float* wproj = (float*)carve(ND * 4);
  float* wx1   = (float*)carve(ND * 4);
  float* wx2   = (float*)carve(ND * 4);
  u16* xbf   = (u16*)carve(ND * 2);
  u16* encbf = (u16*)carve(ND * 2);
  u16* qbf   = (u16*)carve(ND * 2);
  u16* kbf   = (u16*)carve(ND * 2);
  u16* vtbf  = (u16*)carve((size_t)BATCH * HEADS * DEPTH * SEQ * 2);
  u16* ctxbf = (u16*)carve(ND * 2);
  u16* x1bf  = (u16*)carve(ND * 2);
  u16* x2bf  = (u16*)carve(ND * 2);
  u16* hbf   = (u16*)carve((size_t)NROWS * FFN * 2);
  u16* wrel  = (u16*)carve((size_t)NROWS * HEADS * RELP * 2);      // 69 MB
  u16* wabf  = (u16*)carve((size_t)BATCH * HEADS * SEQ * SEQ * 2); // 134 MB
  u16* sWqt = (u16*)carve((size_t)D_MODEL*D_MODEL*2);
  u16* sWkt = (u16*)carve((size_t)D_MODEL*D_MODEL*2);
  u16* sWvt = (u16*)carve((size_t)D_MODEL*D_MODEL*2);
  u16* sWot = (u16*)carve((size_t)D_MODEL*D_MODEL*2);
  u16* cWqt = (u16*)carve((size_t)D_MODEL*D_MODEL*2);
  u16* cWkt = (u16*)carve((size_t)D_MODEL*D_MODEL*2);
  u16* cWvt = (u16*)carve((size_t)D_MODEL*D_MODEL*2);
  u16* cWot = (u16*)carve((size_t)D_MODEL*D_MODEL*2);
  u16* W1t  = (u16*)carve((size_t)D_MODEL*FFN*2);
  u16* W2t  = (u16*)carve((size_t)D_MODEL*FFN*2);
  u16* srel = (u16*)carve((size_t)RELN*DEPTH*2);
  u16* crel = (u16*)carve((size_t)RELN*DEPTH*2);

  const int D = D_MODEL;
  const int nRows = BATCH * HEADS * SEQ;

  // ---- one-shot conversions ----
  {
    int n4 = (int)(ND / 4);
    cvt_bf16<<<dim3((n4+255)/256), dim3(256), 0, stream>>>(inputs, xbf, n4);
    cvt_bf16<<<dim3((n4+255)/256), dim3(256), 0, stream>>>(enc_kv, encbf, n4);
    int r4 = RELN*DEPTH/4;
    cvt_bf16<<<dim3((r4+255)/256), dim3(256), 0, stream>>>(sa_rel, srel, r4);
    cvt_bf16<<<dim3((r4+255)/256), dim3(256), 0, stream>>>(ca_rel, crel, r4);
    dim3 gdd(D/32, D/32);
    cvt_t_bf16<<<gdd, 256, 0, stream>>>(sa_Wq, sWqt, D, D);
    cvt_t_bf16<<<gdd, 256, 0, stream>>>(sa_Wk, sWkt, D, D);
    cvt_t_bf16<<<gdd, 256, 0, stream>>>(sa_Wv, sWvt, D, D);
    cvt_t_bf16<<<gdd, 256, 0, stream>>>(sa_Wo, sWot, D, D);
    cvt_t_bf16<<<gdd, 256, 0, stream>>>(ca_Wq, cWqt, D, D);
    cvt_t_bf16<<<gdd, 256, 0, stream>>>(ca_Wk, cWkt, D, D);
    cvt_t_bf16<<<gdd, 256, 0, stream>>>(ca_Wv, cWvt, D, D);
    cvt_t_bf16<<<gdd, 256, 0, stream>>>(ca_Wo, cWot, D, D);
    cvt_t_bf16<<<dim3(FFN/32, D/32), 256, 0, stream>>>(W1, W1t, D, FFN);
    cvt_t_bf16<<<dim3(D/32, FFN/32), 256, 0, stream>>>(W2, W2t, FFN, D);
  }

  // ================= self-attention =================
  launch_gemm(stream, 0, NROWS, D, D, D, D, D, D, xbf, sWqt, nullptr, qbf, nullptr, sa_bq, nullptr, 0, 0, 0, nullptr, nullptr, 1);
  launch_gemm(stream, 0, NROWS, D, D, D, D, D, D, xbf, sWkt, nullptr, kbf, nullptr, sa_bk, nullptr, 0, 0, 0, nullptr, nullptr, 1);
  launch_gemm(stream, 0, NROWS, D, D, D, D, D, D, xbf, sWvt, nullptr, nullptr, vtbf, sa_bv, nullptr, 0, 0, 0, nullptr, nullptr, 1);
  // rel_q[m,h,r] = Q_head @ rel_emb^T  (z = head; rel already [N=513][K=64])
  launch_gemm(stream, 0, NROWS, RELP, RELN, DEPTH, D, DEPTH, HEADS*RELP,
              qbf, srel, nullptr, wrel, nullptr, nullptr, nullptr, 0, DEPTH, RELP, nullptr, nullptr, HEADS);
  launch_gemm(stream, 1, SEQ, SEQ, SEQ, DEPTH, D, D, SEQ,
              qbf, kbf, attnw_sa, nullptr, nullptr, nullptr, nullptr, 0, 0, 0, wrel, look_mask, BATCH*HEADS);
  softmax1024<<<dim3(nRows), dim3(256), 0, stream>>>(attnw_sa, wabf);
  launch_gemm(stream, 2, SEQ, DEPTH, DEPTH, SEQ, SEQ, SEQ, D,
              wabf, vtbf, nullptr, ctxbf, nullptr, nullptr, nullptr, 0, 0, 0, nullptr, nullptr, BATCH*HEADS);
  launch_gemm(stream, 0, NROWS, D, D, D, D, D, D, ctxbf, sWot, wproj, nullptr, nullptr, sa_bo, inputs, 0, 0, 0, nullptr, nullptr, 1);
  layernorm512<<<dim3(NROWS), dim3(256), 0, stream>>>(wproj, sa_g, sa_b, wx1, x1bf);

  // ================= cross-attention =================
  launch_gemm(stream, 0, NROWS, D, D, D, D, D, D, x1bf,  cWqt, nullptr, qbf, nullptr, ca_bq, nullptr, 0, 0, 0, nullptr, nullptr, 1);
  launch_gemm(stream, 0, NROWS, D, D, D, D, D, D, encbf, cWkt, nullptr, kbf, nullptr, ca_bk, nullptr, 0, 0, 0, nullptr, nullptr, 1);
  launch_gemm(stream, 0, NROWS, D, D, D, D, D, D, encbf, cWvt, nullptr, nullptr, vtbf, ca_bv, nullptr, 0, 0, 0, nullptr, nullptr, 1);
  launch_gemm(stream, 0, NROWS, RELP, RELN, DEPTH, D, DEPTH, HEADS*RELP,
              qbf, crel, nullptr, wrel, nullptr, nullptr, nullptr, 0, DEPTH, RELP, nullptr, nullptr, HEADS);
  launch_gemm(stream, 1, SEQ, SEQ, SEQ, DEPTH, D, D, SEQ,
              qbf, kbf, attnw_ca, nullptr, nullptr, nullptr, nullptr, 0, 0, 0, wrel, pad_mask, BATCH*HEADS);
  softmax1024<<<dim3(nRows), dim3(256), 0, stream>>>(attnw_ca, wabf);
  launch_gemm(stream, 2, SEQ, DEPTH, DEPTH, SEQ, SEQ, SEQ, D,
              wabf, vtbf, nullptr, ctxbf, nullptr, nullptr, nullptr, 0, 0, 0, nullptr, nullptr, BATCH*HEADS);
  launch_gemm(stream, 0, NROWS, D, D, D, D, D, D, ctxbf, cWot, wproj, nullptr, nullptr, ca_bo, wx1, 0, 0, 0, nullptr, nullptr, 1);
  layernorm512<<<dim3(NROWS), dim3(256), 0, stream>>>(wproj, ca_g, ca_b, wx2, x2bf);

  // ================= FFN =================
  launch_gemm(stream, 0, NROWS, FFN, FFN, D, D, D, FFN,
              x2bf, W1t, nullptr, hbf, nullptr, b1, nullptr, 1, 0, 0, nullptr, nullptr, 1);
  launch_gemm(stream, 0, NROWS, D, D, FFN, FFN, FFN, D,
              hbf, W2t, wproj, nullptr, nullptr, b2, wx2, 0, 0, 0, nullptr, nullptr, 1);
  layernorm512<<<dim3(NROWS), dim3(256), 0, stream>>>(wproj, fg, fb, out, nullptr);
}
